// TissueRelationModule_51737176047904
// MI455X (gfx1250) — compile-verified, matrix operand order checked by reading, not run on hardware
//
#include <hip/hip_runtime.h>
#include <hip/hip_bf16.h>
#include <cstdint>

typedef __attribute__((ext_vector_type(16))) _Float16 v16h;
typedef __attribute__((ext_vector_type(4)))  _Float16 v4h;
typedef __attribute__((ext_vector_type(8)))  float    v8f;

#define BNS 0.9999950000374996f   // 1/sqrt(1+1e-5)

// Build a 16-element f16 WMMA fragment per the CDNA5 16-bit A layout:
// lanes 0-15 hold K = {0..7, 16..23}, lanes 16-31 hold K = {8..15, 24..31}.
__device__ __forceinline__ v16h frag16(const _Float16* row, int koff) {
  v4h a0 = *(const v4h*)(row + koff);
  v4h a1 = *(const v4h*)(row + koff + 4);
  v4h a2 = *(const v4h*)(row + 16 + koff);
  v4h a3 = *(const v4h*)(row + 16 + koff + 4);
  v16h r = {a0[0],a0[1],a0[2],a0[3], a1[0],a1[1],a1[2],a1[3],
            a2[0],a2[1],a2[2],a2[3], a3[0],a3[1],a3[2],a3[3]};
  return r;
}

// ---------------- kernel 0: repack W_rpn [oc][ic][3][3] f32 -> [tap][oc][ic] f16
__global__ void repack_wrpn(const float* __restrict__ Wrpn, _Float16* __restrict__ Wh) {
  int idx = blockIdx.x * 256 + threadIdx.x;
  if (idx >= 9 * 128 * 256) return;
  int t   = idx / (128 * 256);
  int rem = idx - t * (128 * 256);
  int oc  = rem >> 8;
  int ic  = rem & 255;
  Wh[idx] = (_Float16)Wrpn[(oc * 256 + ic) * 9 + t];
}

// ---------------- kernel 1: 3x3 maxpool + threshold -> f16 masks, per-class pixel counts
__global__ void __launch_bounds__(128) mask_kernel(const float* __restrict__ seg,
                                                   _Float16* __restrict__ masks,
                                                   float* __restrict__ csum) {
  int blk = blockIdx.x;          // (b*12 + c)*128 + y
  int y   = blk & 127;
  int bc  = blk >> 7;
  int x   = threadIdx.x;
  const float* base = seg + (size_t)bc * 16384;
  float m = -3.4e38f;
  for (int dy = -1; dy <= 1; ++dy) {
    int ry = y + dy;
    if (ry < 0 || ry > 127) continue;
    const float* r = base + ry * 128;
    for (int dx = -1; dx <= 1; ++dx) {
      int rx = x + dx;
      if (rx < 0 || rx > 127) continue;
      m = fmaxf(m, r[rx]);
    }
  }
  float v = (m > 0.5f) ? 1.0f : 0.0f;
  masks[(size_t)bc * 16384 + y * 128 + x] = (_Float16)v;
  unsigned long long bal = __ballot(v > 0.0f);   // wave32: low 32 bits
  if ((threadIdx.x & 31) == 0) {
    int cnt = __popcll(bal);
    if (cnt) atomicAdd(&csum[bc], (float)cnt);
  }
}

// ---------------- kernel 2: implicit-GEMM 3x3 conv via WMMA, fused bias+BN+ReLU
// Block = one (b,y) row: 128 px x 128 oc. Feature halo rows staged per ic-chunk;
// per-tap B tiles double-buffered in LDS (staged once per block, not per wave).
__global__ void __launch_bounds__(256) conv_rpn(const float* __restrict__ feat,
                                                const _Float16* __restrict__ Wh,
                                                const float* __restrict__ brpn,
                                                _Float16* __restrict__ refined) {
  __shared__ _Float16 A3[3 * 130 * 36];    // [row][x+1][ic], padded ic-stride 36
  __shared__ _Float16 Bb[2][128 * 36];     // double-buffered [oc][ic], stride 36
  int blk = blockIdx.x;                    // b*128 + y
  int b = blk >> 7, y = blk & 127;
  int tid  = threadIdx.x;
  int lane = tid & 31;
  int wv   = tid >> 5;
  int l15  = lane & 15;
  int koff = (lane < 16) ? 0 : 8;
  int m0   = wv * 16;

  v8f acc[8];
#pragma unroll
  for (int n = 0; n < 8; ++n) acc[n] = 0.0f;

  int ici = tid >> 3;          // ic within chunk: 0..31
  int xg  = (tid & 7) * 16;    // 16-wide x group

  // cooperative B-tile stage: 256 threads x 16 f16 (32B coalesced reads)
  int soc = tid >> 1;
  int skh = (tid & 1) * 16;
  auto stageB = [&](int tap, int icc, _Float16* dst) {
    const v4h* src = (const v4h*)(Wh + ((size_t)(tap * 128 + soc)) * 256 + icc + skh);
    v4h t0 = src[0], t1 = src[1], t2 = src[2], t3 = src[3];
    _Float16* d = dst + soc * 36 + skh;
    *(v4h*)(d)      = t0;
    *(v4h*)(d + 4)  = t1;
    *(v4h*)(d + 8)  = t2;
    *(v4h*)(d + 12) = t3;
  };

  stageB(0, 0, Bb[0]);        // prologue: tap 0, chunk 0
  int cb = 0;

  for (int ic0 = 0; ic0 < 256; ic0 += 32) {
    // ---- stage rows y-1, y, y+1 for this ic chunk (row validity is block-uniform)
#pragma unroll
    for (int r = 0; r < 3; ++r) {
      int ry = y + r - 1;
      if (ry >= 0 && ry < 128) {
        const float* srow =
            feat + ((size_t)(b * 256 + ic0 + ici) * 128 + ry) * 128 + xg;
#pragma unroll
        for (int j = 0; j < 16; ++j)
          A3[(r * 130 + xg + j + 1) * 36 + ici] = (_Float16)srow[j];
      } else {
#pragma unroll
        for (int j = 0; j < 16; ++j)
          A3[(r * 130 + xg + j + 1) * 36 + ici] = (_Float16)0.0f;
      }
      if (xg == 0)   A3[(r * 130 + 0)   * 36 + ici] = (_Float16)0.0f;
      if (xg == 112) A3[(r * 130 + 129) * 36 + ici] = (_Float16)0.0f;
    }
    // prefetch next chunk's center row (lowers to global_prefetch_b8)
    if (ic0 + 32 < 256) {
      const float* nxt =
          feat + ((size_t)(b * 256 + ic0 + 32 + ici) * 128 + y) * 128 + xg;
      __builtin_prefetch(nxt, 0, 1);
    }
    __syncthreads();
    // ---- 9 taps x 8 oc-tiles of WMMA; stage next tap's B while computing
    for (int t9 = 0; t9 < 9; ++t9) {
      int nt9 = t9 + 1, nic0 = ic0;
      if (nt9 == 9) { nt9 = 0; nic0 += 32; }
      if (nic0 < 256) stageB(nt9, nic0, Bb[cb ^ 1]);   // scalar-uniform guard

      int dy = t9 / 3, dx = t9 % 3;      // 0..2 each
      v16h a = frag16(&A3[(dy * 130 + m0 + l15 + dx) * 36], koff);
      const _Float16* brow = &Bb[cb][l15 * 36];
#pragma unroll
      for (int n = 0; n < 8; ++n) {
        v16h bf = frag16(brow + n * 16 * 36, koff);
        acc[n] = __builtin_amdgcn_wmma_f32_16x16x32_f16(
            false, a, false, bf, (short)0, acc[n], false, false);
      }
      cb ^= 1;
      __syncthreads();
    }
  }
  // epilogue: C layout -> VGPR r: lanes 0-15 M=r, lanes 16-31 M=8+r; N=lane&15
  int hi8 = (lane < 16) ? 0 : 8;
#pragma unroll
  for (int n = 0; n < 8; ++n) {
    int oc = n * 16 + l15;
    float bias = brpn[oc];
#pragma unroll
    for (int r = 0; r < 8; ++r) {
      int pix = m0 + r + hi8;
      float v = fmaxf((acc[n][r] + bias) * BNS, 0.0f);
      refined[((size_t)blk * 128 + pix) * 128 + oc] = (_Float16)v;
    }
  }
}

// ---------------- kernel 3: 1x1 conv (WMMA, K=128 N=64) + masked per-class sums
__global__ void __launch_bounds__(256) node_feat(const _Float16* __restrict__ refined,
                                                 const float* __restrict__ Wnode,
                                                 const float* __restrict__ bnode,
                                                 const _Float16* __restrict__ masks,
                                                 float* __restrict__ msum) {
  __shared__ _Float16 Bh[64 * 132];    // W_node[n][k] f16, padded stride 132
  __shared__ float    zf[128 * 64];    // z for the 128 pixels of this row
  __shared__ _Float16 mrow[12 * 128];  // mask row per class
  int blk = blockIdx.x; int b = blk >> 7; int y = blk & 127;
  int tid = threadIdx.x;
  {
    int n = tid >> 2, q = (tid & 3) * 32;
    const float* src = Wnode + n * 128 + q;
#pragma unroll
    for (int j = 0; j < 32; ++j) Bh[n * 132 + q + j] = (_Float16)src[j];
  }
  for (int idx = tid; idx < 12 * 128; idx += 256) {
    int c = idx >> 7, x = idx & 127;
    mrow[idx] = masks[((size_t)(b * 12 + c)) * 16384 + y * 128 + x];
  }
  __syncthreads();
  int lane = tid & 31, wv = tid >> 5;
  int l15 = lane & 15, koff = (lane < 16) ? 0 : 8, m0 = wv * 16;
  v8f acc[4];
#pragma unroll
  for (int n = 0; n < 4; ++n) acc[n] = 0.0f;
  const _Float16* arow = refined + ((size_t)blk * 128 + (m0 + l15)) * 128;
#pragma unroll
  for (int kc = 0; kc < 4; ++kc) {
    v16h a = frag16(arow + kc * 32, koff);
#pragma unroll
    for (int n = 0; n < 4; ++n) {
      v16h bf = frag16(&Bh[(n * 16 + l15) * 132 + kc * 32], koff);
      acc[n] = __builtin_amdgcn_wmma_f32_16x16x32_f16(
          false, a, false, bf, (short)0, acc[n], false, false);
    }
  }
  int hi8 = (lane < 16) ? 0 : 8;
#pragma unroll
  for (int n = 0; n < 4; ++n) {
    int hd = n * 16 + l15;
    float bias = bnode[hd];
#pragma unroll
    for (int r = 0; r < 8; ++r) {
      int pix = m0 + r + hi8;
      zf[pix * 64 + hd] = fmaxf((acc[n][r] + bias) * BNS, 0.0f);
    }
  }
  __syncthreads();
  if (tid < 64) {
    int hd = tid;
    for (int c = 0; c < 12; ++c) {
      float s = 0.0f;
      for (int p = 0; p < 128; ++p)
        s += zf[p * 64 + hd] * (float)mrow[c * 128 + p];
      atomicAdd(&msum[((size_t)(b * 12 + c)) * 64 + hd], s);
    }
  }
}

// ---------------- kernel 4: tiny graph network (one block per batch)
__global__ void __launch_bounds__(128) graph_net(
    const float* __restrict__ msum, const float* __restrict__ csum,
    const float* __restrict__ We1, const float* __restrict__ be1,
    const float* __restrict__ We2, const float* __restrict__ be2,
    const float* __restrict__ Wg,  const float* __restrict__ bg,
    const float* __restrict__ lng, const float* __restrict__ lnb,
    const float* __restrict__ gemb, const float* __restrict__ Wsp,
    float* __restrict__ ytab) {
  __shared__ float nf[12 * 64], xx[12 * 64], xw[12 * 64], xr[12 * 64];
  __shared__ float ea[132 * 64];
  __shared__ float hbuf[2 * 64];
  __shared__ float present[12], presence[12], mu[12], rs[12];
  int b = blockIdx.x, t = threadIdx.x;
  for (int idx = t; idx < 768; idx += 128) {
    int c = idx >> 6;
    float ms = csum[b * 12 + c];
    nf[idx] = msum[(b * 12 + c) * 64 + (idx & 63)] / (ms + 1e-6f);
  }
  __syncthreads();
  if (t < 12) {
    present[t] = (csum[b * 12 + t] > 0.0f) ? 1.0f : 0.0f;
    float s = 0.0f;
    for (int d = 0; d < 64; ++d) s += nf[t * 64 + d];
    presence[t] = (s != 0.0f) ? 1.0f : 0.0f;
  }
  __syncthreads();
  // edges: e = i*11 + (j if j<i else j-1); 2 edges in flight across 128 threads
  int sub = t >> 6, k = t & 63;
  for (int e0 = 0; e0 < 132; e0 += 2) {
    int e = e0 + sub;
    int i = e / 11, jj = e % 11;
    int j = jj + (jj >= i ? 1 : 0);
    float h = be1[k];
    for (int q = 0; q < 64; ++q) {
      h += nf[i * 64 + q] * We1[q * 64 + k];
      h += nf[j * 64 + q] * We1[(64 + q) * 64 + k];
    }
    hbuf[sub * 64 + k] = fmaxf(h, 0.0f);
    __syncthreads();
    float v = be2[k];
    for (int q = 0; q < 64; ++q) v += hbuf[sub * 64 + q] * We2[q * 64 + k];
    ea[e * 64 + k] = v * present[i] * present[j];
    __syncthreads();
  }
  for (int idx = t; idx < 768; idx += 128) xx[idx] = nf[idx];
  __syncthreads();
  for (int l = 0; l < 3; ++l) {
    for (int idx = t; idx < 768; idx += 128) {
      int c = idx >> 6, d = idx & 63;
      float a = 0.0f;
      for (int q = 0; q < 64; ++q) a += xx[c * 64 + q] * Wg[(l * 64 + q) * 64 + d];
      xw[idx] = a;
    }
    __syncthreads();
    for (int idx = t; idx < 768; idx += 128) {
      int j = idx >> 6, d = idx & 63;
      float a = bg[l * 64 + d];
      for (int i2 = 0; i2 < 12; ++i2) {
        if (i2 == j) continue;
        int e = i2 * 11 + (j > i2 ? j - 1 : j);
        a += xw[i2 * 64 + d] * ea[e * 64 + d];
      }
      xr[idx] = fmaxf(a, 0.0f);
    }
    __syncthreads();
    if (t < 12) {
      float m = 0.0f;
      for (int d = 0; d < 64; ++d) m += xr[t * 64 + d];
      m *= (1.0f / 64.0f);
      float v = 0.0f;
      for (int d = 0; d < 64; ++d) { float dd = xr[t * 64 + d] - m; v += dd * dd; }
      v *= (1.0f / 64.0f);
      mu[t] = m;
      rs[t] = rsqrtf(v + 1e-5f);
    }
    __syncthreads();
    for (int idx = t; idx < 768; idx += 128) {
      int c = idx >> 6, d = idx & 63;
      float nm = (xr[idx] - mu[c]) * rs[c] * lng[l * 64 + d] + lnb[l * 64 + d];
      xx[idx] = (l > 0) ? (nm + xx[idx]) : nm;
    }
    __syncthreads();
  }
  for (int idx = t; idx < 768; idx += 128) {
    int c = idx >> 6;
    float p = presence[c];
    xx[idx] = xx[idx] * p + gemb[idx] * (1.0f - p);
  }
  __syncthreads();
  // fold x @ W_sp -> per-(b,c) 128-wide rows for the spatial projector
  for (int idx = t; idx < 12 * 128; idx += 128) {
    int c = idx >> 7, o = idx & 127;
    float a = 0.0f;
    for (int d = 0; d < 64; ++d) a += xx[c * 64 + d] * Wsp[d * 128 + o];
    ytab[(b * 12 + c) * 128 + o] = a;
  }
}

// ---------------- kernel 5: spatial projector + bias/BN/ReLU (bandwidth bound)
__global__ void __launch_bounds__(256) spatial_proj(
    const float* __restrict__ ytab, const _Float16* __restrict__ masks,
    const float* __restrict__ bsp, float* __restrict__ out) {
  __shared__ float    yt[12 * 128];
  __shared__ _Float16 mr[12 * 128];
  int blk = blockIdx.x; int b = blk >> 7; int y = blk & 127;
  int t = threadIdx.x;
  for (int idx = t; idx < 1536; idx += 256) {
    yt[idx] = ytab[(size_t)b * 1536 + idx];
    int c = idx >> 7, x = idx & 127;
    mr[idx] = masks[((size_t)(b * 12 + c)) * 16384 + y * 128 + x];
  }
  __syncthreads();
  int x = t & 127, half = (t >> 7) * 64;
  float acc[64];
#pragma unroll
  for (int o = 0; o < 64; ++o) acc[o] = bsp[half + o];
  for (int c = 0; c < 12; ++c) {
    float mv = (float)mr[c * 128 + x];
#pragma unroll
    for (int o = 0; o < 64; ++o) acc[o] += mv * yt[c * 128 + half + o];
  }
#pragma unroll
  for (int o = 0; o < 64; ++o) {
    int oc = half + o;
    out[((size_t)(b * 128 + oc)) * 16384 + y * 128 + x] = fmaxf(acc[o] * BNS, 0.0f);
  }
}

extern "C" void kernel_launch(void* const* d_in, const int* in_sizes, int n_in,
                              void* d_out, int out_size, void* d_ws, size_t ws_size,
                              hipStream_t stream) {
  const float* feat  = (const float*)d_in[0];
  const float* seg   = (const float*)d_in[1];
  const float* Wrpn  = (const float*)d_in[2];
  const float* brpn  = (const float*)d_in[3];
  const float* Wnode = (const float*)d_in[4];
  const float* bnode = (const float*)d_in[5];
  const float* We1   = (const float*)d_in[6];
  const float* be1   = (const float*)d_in[7];
  const float* We2   = (const float*)d_in[8];
  const float* be2   = (const float*)d_in[9];
  const float* Wg    = (const float*)d_in[10];
  const float* bg    = (const float*)d_in[11];
  const float* lng   = (const float*)d_in[12];
  const float* lnb   = (const float*)d_in[13];
  const float* gemb  = (const float*)d_in[14];
  const float* Wsp   = (const float*)d_in[15];
  const float* bsp   = (const float*)d_in[16];
  float* out = (float*)d_out;

  uint8_t* ws = (uint8_t*)d_ws;
  size_t off = 0;
  _Float16* refined = (_Float16*)(ws + off); off += 33554432;  // 131072*128 f16
  _Float16* Wh      = (_Float16*)(ws + off); off += 589824;    // 9*128*256 f16
  _Float16* masks   = (_Float16*)(ws + off); off += 3145728;   // 8*12*16384 f16
  float*    msum    = (float*)(ws + off);    off += 24576;     // 8*12*64 f32
  float*    csum    = (float*)(ws + off);    off += 512;       // 8*12 f32
  float*    ytab    = (float*)(ws + off);    off += 49152;     // 8*12*128 f32

  hipMemsetAsync(msum, 0, 24576, stream);
  hipMemsetAsync(csum, 0, 512, stream);

  repack_wrpn<<<1152, 256, 0, stream>>>(Wrpn, Wh);
  mask_kernel<<<12288, 128, 0, stream>>>(seg, masks, csum);
  conv_rpn<<<1024, 256, 0, stream>>>(feat, Wh, brpn, refined);
  node_feat<<<1024, 256, 0, stream>>>(refined, Wnode, bnode, masks, msum);
  graph_net<<<8, 128, 0, stream>>>(msum, csum, We1, be1, We2, be2,
                                   Wg, bg, lng, lnb, gemb, Wsp, ytab);
  spatial_proj<<<1024, 256, 0, stream>>>(ytab, masks, bsp, out);
}
